// LSTM_2267742732466
// MI455X (gfx1250) — matrix operand order repaired and checked by reading, not compiled, not run on hardware
//
#include <hip/hip_runtime.h>
#include <hip/hip_bf16.h>

// ---------------------------------------------------------------------------
// LSTM on MI455X (gfx1250, wave32, WMMA).
//  Phase 1: x-projection of ALL timesteps as one big bf16 WMMA GEMM
//           (M=32768, K=512, N=2048 = 4 gates fused), bias folded in.
//  Phase 2: ONE persistent kernel runs all 512 recurrent steps:
//           - WhT weights staged once into LDS (128 KB/block), read via ds_load
//           - c-state lives in registers for the whole sequence
//           - h exchanged via double-buffered global array + grid barrier
// ---------------------------------------------------------------------------

typedef __bf16 bf16;
typedef __attribute__((ext_vector_type(16))) __bf16 v16bf;
typedef __attribute__((ext_vector_type(8)))  __bf16 v8bf;
typedef __attribute__((ext_vector_type(8)))  float  v8f;

#define T_STEPS 512
#define BATCH   64
#define EMBD    512
#define LATD    512
#define VOCAB   32000
#define NCAT    2048                 // 4 gates * LATD, gate-major columns
#define MTOT    (T_STEPS * BATCH)    // 32768
#define RGRID   16                   // blocks in persistent recurrent kernel

struct AB2 { v8bf lo, hi; };

static __device__ inline v16bf make16(v8bf lo, v8bf hi) {
    AB2 t{lo, hi};
    return __builtin_bit_cast(v16bf, t);
}

// A fragment: 16x32 bf16 tile, row-major source. rowptr already points at
// (row m_of_this_lane, k0). Layout: lanes 0-15 hold K=0..15,
// lanes 16-31 hold K=16..31.
static __device__ inline v16bf load_a(const bf16* rowptr, int lane) {
    const int kb = (lane >> 4) << 4;              // 0 or 16
    v8bf lo = *(const v8bf*)(rowptr + kb);        // K = kb .. kb+7
    v8bf hi = *(const v8bf*)(rowptr + kb + 8);    // K = kb+8 .. kb+15
    return make16(lo, hi);
}

// B fragment: 32x16 bf16 tile from PRE-TRANSPOSED weights WT[N][K] so each
// lane reads 16 contiguous K values: lane n<16 -> K=0..15, n>=16 -> K=16..31.
static __device__ inline v16bf load_b(const bf16* wt, int ldk, int n0, int k0, int lane) {
    const int n  = lane & 15;
    const int kb = (lane >> 4) << 4;              // 0 or 16
    const bf16* p = wt + (size_t)(n0 + n) * ldk + k0 + kb;
    v8bf lo = *(const v8bf*)(p);
    v8bf hi = *(const v8bf*)(p + 8);
    return make16(lo, hi);
}

static __device__ inline v8f wmma_bf16(v16bf a, v16bf b, v8f c) {
    return __builtin_amdgcn_wmma_f32_16x16x32_bf16(
        /*neg_a=*/false, a, /*neg_b=*/false, b,
        /*c_mod=*/(short)0, c, /*reuse_a=*/false, /*reuse_b=*/false);
}

static __device__ inline float sigmoidf_fast(float x) {
    return 1.0f / (1.0f + __expf(-x));
}

// --------------------------- prep kernels ----------------------------------

__global__ void __launch_bounds__(256)
cvt_bf16_kernel(const float* __restrict__ src, bf16* __restrict__ dst, int n) {
    int i = blockIdx.x * blockDim.x + threadIdx.x;
    int stride = gridDim.x * blockDim.x;
    for (; i < n; i += stride) dst[i] = (bf16)src[i];
}

// Build gate-major, transposed bf16 weights:
//   WxT[g*512+n][k] = Wg[k      ][n]   (x-part rows 0..511 of Wg)
//   WhT[g*512+n][k] = Wg[512 + k][n]   (h-part rows 512..1023)
// and biasCat[g*512+n] = bg[n].
__global__ void __launch_bounds__(256)
pack_w_kernel(const float* __restrict__ Wf, const float* __restrict__ Wi,
              const float* __restrict__ Wc, const float* __restrict__ Wo,
              const float* __restrict__ bfp, const float* __restrict__ bip,
              const float* __restrict__ bcp, const float* __restrict__ bop,
              bf16* __restrict__ WxT, bf16* __restrict__ WhT,
              float* __restrict__ biasCat) {
    int i = blockIdx.x * blockDim.x + threadIdx.x;   // over NCAT*EMBD
    if (i >= NCAT * EMBD) return;
    int ncat = i / EMBD;
    int k    = i % EMBD;
    int g    = ncat / LATD;
    int n    = ncat % LATD;
    const float* W = (g == 0) ? Wf : (g == 1) ? Wi : (g == 2) ? Wc : Wo;
    WxT[i] = (bf16)W[(size_t)k * LATD + n];
    WhT[i] = (bf16)W[(size_t)(EMBD + k) * LATD + n];
    if (k == 0) {
        const float* bb = (g == 0) ? bfp : (g == 1) ? bip : (g == 2) ? bcp : bop;
        biasCat[ncat] = bb[n];
    }
}

__global__ void __launch_bounds__(256)
init_state_kernel(const float* __restrict__ h0, bf16* __restrict__ hbuf,
                  int* __restrict__ bar, int n) {
    int i = blockIdx.x * blockDim.x + threadIdx.x;
    if (i >= n) return;
    hbuf[i] = (bf16)h0[i];
    if (i == 0) *bar = 0;     // grid-barrier counter: reset every launch
}

// ----------------- phase 1: x-projection GEMM (all timesteps) --------------
// Wave tile: 32 rows (M) x 64 cols (N). 8 accumulators, 2 A frags shared
// across 4 B frags per K chunk -> 8 v_wmma per iteration, 16 iterations.
__global__ void __launch_bounds__(256)
xproj_gemm_kernel(const int*  __restrict__ x,        // [MTOT] token ids
                  const bf16* __restrict__ embb,     // [VOCAB, EMBD]
                  const bf16* __restrict__ WxT,      // [NCAT, EMBD]
                  const float* __restrict__ biasCat, // [NCAT]
                  bf16* __restrict__ Xg)             // [MTOT, NCAT]
{
    const int lane = threadIdx.x & 31;
    const int warp = threadIdx.x >> 5;
    const int w    = blockIdx.x * 8 + warp;          // 0..32767
    const int NSLAB = NCAT / 64;                     // 32
    const int mt = w / NSLAB;                        // 0..1023 (32 rows each)
    const int ns = w % NSLAB;
    const int m0 = mt * 32;
    const int n0 = ns * 64;

    const int mlane = lane & 15;
    const bf16* arow0 = embb + (size_t)x[m0 + mlane]      * EMBD;  // gather
    const bf16* arow1 = embb + (size_t)x[m0 + 16 + mlane] * EMBD;

    v8f acc[2][4] = {};
    for (int k0 = 0; k0 < EMBD; k0 += 32) {
        v16bf a0 = load_a(arow0 + k0, lane);
        v16bf a1 = load_a(arow1 + k0, lane);
#pragma unroll
        for (int j = 0; j < 4; ++j) {
            v16bf b = load_b(WxT, EMBD, n0 + j * 16, k0, lane);
            acc[0][j] = wmma_bf16(a0, b, acc[0][j]);
            acc[1][j] = wmma_bf16(a1, b, acc[1][j]);
        }
    }

    const int nlane = lane & 15;
    const int mo    = (lane >> 4) * 8;
#pragma unroll
    for (int i = 0; i < 2; ++i) {
#pragma unroll
        for (int j = 0; j < 4; ++j) {
            const int ncol = n0 + j * 16 + nlane;
            const float bia = biasCat[ncol];
#pragma unroll
            for (int r = 0; r < 8; ++r) {
                const int row = m0 + i * 16 + mo + r;
                Xg[(size_t)row * NCAT + ncol] = (bf16)(acc[i][j][r] + bia);
            }
        }
    }
}

// ----------------- phase 2: persistent recurrent kernel --------------------
// 16 blocks x 8 waves = 128 waves. Wave w: jt = w>>2 (0..31), bt = w&3 (0..3)
// so each block's 8 waves need only 2 weight slices (2 x 64 KB) -> LDS.
// Per step per wave: 16 K-chunks x 4 gates = 64 v_wmma, B frags from LDS,
// c-state in registers for the entire sequence, grid barrier between steps.

__global__ void __launch_bounds__(256)
lstm_recurrent_kernel(const bf16* __restrict__ WhT,  // [NCAT, LATD] (global)
                      const bf16* __restrict__ Xg,   // [MTOT, NCAT]
                      bf16* __restrict__ hbuf,       // [2][BATCH*LATD] bf16
                      float* __restrict__ hs,        // [T, BATCH, LATD] f32
                      int* __restrict__ bar)         // grid barrier counter
{
    extern __shared__ bf16 wlds[];                   // [2][4][16][LATD] = 128KB

    const int lane = threadIdx.x & 31;
    const int warp = threadIdx.x >> 5;
    const int w    = blockIdx.x * 8 + warp;          // 0..127
    const int jt   = w >> 2;                         // 0..31
    const int bt   = w & 3;                          // 0..3
    const int sl   = warp >> 2;                      // slice in LDS: 0/1
    const int b0 = bt * 16, j0 = jt * 16;

    // ---- one-time LDS fill: 2 slices x (4 gates x 16 cols x 512 K) bf16 ----
    {
        const int jtBase = blockIdx.x * 2;           // first jt of this block
        // 2*4*16*512 bf16 = 65536 elements = 8192 uint4 (16B) chunks
        const uint4* src0 = (const uint4*)WhT;
        uint4* dst = (uint4*)wlds;
        for (int idx = threadIdx.x; idx < 8192; idx += 256) {
            // dst element idx -> (s, g, nloc, kchunk) ; 64 chunks per row
            const int kc   = idx & 63;
            const int row  = idx >> 6;               // 0..127 = s*64 + g*16 + nloc
            const int s    = row >> 6;
            const int g    = (row >> 4) & 3;
            const int nloc = row & 15;
            const int grow = g * LATD + (jtBase + s) * 16 + nloc; // row in WhT
            dst[idx] = src0[(size_t)grow * (LATD / 8) + kc];
        }
    }
    __syncthreads();

    const bf16* wl = wlds + (size_t)sl * (4 * 16 * LATD);

    const int mlane = lane & 15;
    const int n     = lane & 15;
    const int mo    = (lane >> 4) * 8;
    const int j     = j0 + n;

    float creg[8] = {0.f, 0.f, 0.f, 0.f, 0.f, 0.f, 0.f, 0.f};  // c0 = zeros
    int   done = 0;                                  // monotonic barrier count

    for (int t = 0; t < T_STEPS; ++t) {
        const bf16* hin  = hbuf + (size_t)(t & 1) * (BATCH * LATD);
        bf16*       hout = hbuf + (size_t)((t + 1) & 1) * (BATCH * LATD);
        const bf16* arow = hin + (size_t)(b0 + mlane) * LATD;

        v8f acc[4] = {};
        for (int k0 = 0; k0 < LATD; k0 += 32) {
            v16bf a = load_a(arow + k0, lane);
#pragma unroll
            for (int g = 0; g < 4; ++g) {
                // B fragment from LDS (ds_load): row = g*16 + nloc, 16 bf16
                const int kb = (lane >> 4) << 4;
                const bf16* p = wl + ((g * 16 + n) * LATD) + k0 + kb;
                v8bf lo = *(const v8bf*)(p);
                v8bf hi = *(const v8bf*)(p + 8);
                acc[g] = wmma_bf16(a, make16(lo, hi), acc[g]);
            }
        }

#pragma unroll
        for (int r = 0; r < 8; ++r) {
            const int b = b0 + mo + r;
            const size_t xrow = (size_t)(t * BATCH + b) * NCAT;
            const float pf = acc[0][r] + (float)Xg[xrow + 0 * LATD + j];
            const float pi = acc[1][r] + (float)Xg[xrow + 1 * LATD + j];
            const float pc = acc[2][r] + (float)Xg[xrow + 2 * LATD + j];
            const float po = acc[3][r] + (float)Xg[xrow + 3 * LATD + j];
            const float fg = sigmoidf_fast(pf);
            const float ig = sigmoidf_fast(pi);
            const float gg = tanhf(pc);
            const float og = sigmoidf_fast(po);
            const float cn = fg * creg[r] + ig * gg;
            creg[r] = cn;
            const float hn = og * tanhf(cn);
            const size_t ci = (size_t)b * LATD + j;
            hs[(size_t)t * (BATCH * LATD) + ci] = hn;
            hout[ci] = (bf16)hn;
        }

        // ---- grid barrier: release h(t+1), wait for all 16 blocks ----
        done += RGRID;
        __threadfence();                 // release this thread's h stores
        __syncthreads();
        if (threadIdx.x == 0) {
            __hip_atomic_fetch_add(bar, 1, __ATOMIC_RELEASE,
                                   __HIP_MEMORY_SCOPE_AGENT);
            while (__hip_atomic_load(bar, __ATOMIC_ACQUIRE,
                                     __HIP_MEMORY_SCOPE_AGENT) < done) {
                __builtin_amdgcn_s_sleep(1);
            }
        }
        __syncthreads();
        __threadfence();                 // acquire other blocks' h stores
    }
}

// ---------------------------------------------------------------------------

extern "C" void kernel_launch(void* const* d_in, const int* in_sizes, int n_in,
                              void* d_out, int out_size, void* d_ws, size_t ws_size,
                              hipStream_t stream) {
    const int*   x   = (const int*)  d_in[0];
    const float* h0  = (const float*)d_in[1];
    const float* emb = (const float*)d_in[2];
    const float* Wf  = (const float*)d_in[3];
    const float* bfp = (const float*)d_in[4];
    const float* Wi  = (const float*)d_in[5];
    const float* bip = (const float*)d_in[6];
    const float* Wc  = (const float*)d_in[7];
    const float* bcp = (const float*)d_in[8];
    const float* Wo  = (const float*)d_in[9];
    const float* bop = (const float*)d_in[10];
    float* hs = (float*)d_out;

    // workspace carve-out (~172 MB total)
    char* p = (char*)d_ws;
    auto alloc = [&](size_t bytes) -> char* {
        char* r = p;
        p += (bytes + 255) & ~(size_t)255;
        return r;
    };
    bf16*  embb  = (bf16*) alloc((size_t)VOCAB * EMBD * sizeof(bf16));
    bf16*  WxT   = (bf16*) alloc((size_t)NCAT * EMBD * sizeof(bf16));
    bf16*  WhT   = (bf16*) alloc((size_t)NCAT * LATD * sizeof(bf16));
    float* biasC = (float*)alloc((size_t)NCAT * sizeof(float));
    bf16*  hbuf  = (bf16*) alloc((size_t)2 * BATCH * LATD * sizeof(bf16));
    int*   bar   = (int*)  alloc(256);
    bf16*  Xg    = (bf16*) alloc((size_t)MTOT * NCAT * sizeof(bf16));

    // prep: f32 -> bf16 conversions, weight transpose/pack, state init
    {
        const int n = VOCAB * EMBD;
        cvt_bf16_kernel<<<(n + 255) / 256, 256, 0, stream>>>(emb, embb, n);
    }
    pack_w_kernel<<<(NCAT * EMBD + 255) / 256, 256, 0, stream>>>(
        Wf, Wi, Wc, Wo, bfp, bip, bcp, bop, WxT, WhT, biasC);
    init_state_kernel<<<(BATCH * LATD + 255) / 256, 256, 0, stream>>>(
        h0, hbuf, bar, BATCH * LATD);

    // phase 1: big parallel GEMM (32768 waves, 8 waves/block)
    xproj_gemm_kernel<<<4096, 256, 0, stream>>>(x, embb, WxT, biasC, Xg);

    // phase 2: ONE persistent kernel for the whole recurrence
    //          (16 blocks co-resident, 128 KB dynamic LDS each)
    lstm_recurrent_kernel<<<RGRID, 256, 2 * 4 * 16 * LATD * sizeof(bf16), stream>>>(
        WhT, Xg, hbuf, hs, bar);
}
